// UltimateMemoryOptimizedModel_12068858102146
// MI455X (gfx1250) — matrix operand order repaired and checked by reading, not compiled, not run on hardware
//
#include <hip/hip_runtime.h>
#include <math.h>

// ---------------------------------------------------------------------------
// MI455X (gfx1250) implementation.
//  - network body: small VALU kernels (memory-bound, ~50 MB total traffic)
//  - logits GEMM: v_wmma_f32_16x16x32_f16, split-f16 3-pass for ~f32 accuracy,
//    table generated on the fly from TT cores (never touches HBM),
//    double-buffered LDS with async global->LDS copies (ASYNCcnt),
//    nontemporal logits stores.
// ---------------------------------------------------------------------------

typedef __attribute__((ext_vector_type(16))) _Float16 v16h;
typedef __attribute__((ext_vector_type(8)))  _Float16 v8h;
typedef __attribute__((ext_vector_type(8)))  float    v8f;

#define DM    1024
#define NLAY  4
#define VOC   50000
#define VF2   200
#define NTOK  2048
#define EPSN  1e-6f
#define LDA   40      // padded LDS row stride (halves): 20 banks -> conflict-free
#define BMT   64      // GEMM block tile M
#define BNT   128     // GEMM block tile N

#if __has_builtin(__builtin_amdgcn_global_load_async_to_lds_b128)
#define HAVE_ASYNC 1
#else
#define HAVE_ASYNC 0
#endif

// builtin signature (from clang diagnostic): (v4i AS1*, v4i AS3*, imm, imm)
typedef int v4i __attribute__((vector_size(16)));
typedef __attribute__((address_space(1))) v4i* gp_v4i;
typedef __attribute__((address_space(3))) v4i* lp_v4i;

// 16-byte global -> LDS copy; async (ASYNCcnt) when available.
__device__ __forceinline__ void g2l_b128(const _Float16* g, _Float16* l) {
#if HAVE_ASYNC
  __builtin_amdgcn_global_load_async_to_lds_b128((gp_v4i)(g), (lp_v4i)(l), 0, 0);
#else
  *(uint4*)l = *(const uint4*)g;
#endif
}

__device__ __forceinline__ void wait_async0() {
#if HAVE_ASYNC
#if __has_builtin(__builtin_amdgcn_s_wait_asynccnt)
  __builtin_amdgcn_s_wait_asynccnt(0);
#else
  asm volatile("s_wait_asynccnt 0x0" ::: "memory");
#endif
#endif
}

// ---------------- helpers ----------------

__device__ __forceinline__ float block_reduce_256(float v, float* sm8) {
  #pragma unroll
  for (int o = 16; o > 0; o >>= 1) v += __shfl_down(v, o, 32);
  __syncthreads();                       // protect sm8 against reuse
  if ((threadIdx.x & 31) == 0) sm8[threadIdx.x >> 5] = v;
  __syncthreads();
  float tot = 0.f;
  #pragma unroll
  for (int i = 0; i < 8; ++i) tot += sm8[i];
  return tot;
}

__device__ __forceinline__ float gelu_tanh(float v) {
  return 0.5f * v * (1.0f + tanhf(0.7978845608028654f * (v + 0.044715f * v * v * v)));
}

// ---------------- 1) HTT embedding: x[tok, d] ----------------

__global__ void k_embed(const int* __restrict__ ids, const float* __restrict__ c1,
                        const float* __restrict__ c2, float* __restrict__ x) {
  const int tok = blockIdx.x;
  const int id  = ids[tok];
  const int i1  = id / VF2;
  const int i2  = id - i1 * VF2;
  const int t   = threadIdx.x;
  #pragma unroll
  for (int j = 0; j < 4; ++j) {
    int d  = t + j * 256;
    int d1 = d >> 5, d2 = d & 31;
    float g0 = c1[(i1 * 32 + d1) * 2 + 0];
    float g1 = c1[(i1 * 32 + d1) * 2 + 1];
    x[tok * DM + d] = g0 * c2[(i2 * 2 + 0) * 32 + d2] + g1 * c2[(i2 * 2 + 1) * 32 + d2];
  }
}

// ---------------- 2) u = rmsnorm(x, nw) @ Bm   (one token per block) --------

__global__ void k_norm_B(const float* __restrict__ x, const float* __restrict__ nw,
                         const float* __restrict__ Bm, float* __restrict__ u) {
  __shared__ float sm[8];
  const int tok = blockIdx.x, t = threadIdx.x;
  float xv[4]; float ss = 0.f;
  #pragma unroll
  for (int j = 0; j < 4; ++j) {
    float v = x[tok * DM + t + j * 256];
    xv[j] = v; ss += v * v;
  }
  float rstd = rsqrtf(block_reduce_256(ss, sm) * (1.0f / DM) + EPSN);
  float acc[4] = {0.f, 0.f, 0.f, 0.f};
  #pragma unroll
  for (int j = 0; j < 4; ++j) {
    int d = t + j * 256;
    float xn = xv[j] * nw[d] * rstd;
    #pragma unroll
    for (int r = 0; r < 4; ++r) acc[r] += xn * Bm[d * 4 + r];
  }
  #pragma unroll
  for (int r = 0; r < 4; ++r) {
    float tot = block_reduce_256(acc[r], sm);
    if (t == 0) u[tok * 4 + r] = tot;
  }
}

// ---------------- 3) sequential semiseparable scan (in place on u) ----------

__global__ void k_scan(float* __restrict__ u, const float* __restrict__ a) {
  const int b = threadIdx.x;
  if (b < 2) {
    const float4 aa = *(const float4*)a;
    float4 h = make_float4(0.f, 0.f, 0.f, 0.f);
    float4* up = (float4*)(u + (size_t)b * 1024 * 4);
    #pragma unroll 8
    for (int t = 0; t < 1024; ++t) {
      float4 ut = up[t];
      h.x = aa.x * h.x + ut.x;
      h.y = aa.y * h.y + ut.y;
      h.z = aa.z * h.z + ut.z;
      h.w = aa.w * h.w + ut.w;
      up[t] = h;
    }
  }
}

// ---------------- 4) x += hs@Cm ; z = gelu(rmsnorm(x,nw)@W1) ----------------

__global__ void k_cy_ffn1(float* __restrict__ x, const float* __restrict__ hs,
                          const float* __restrict__ Cm, const float* __restrict__ nw,
                          const float* __restrict__ W1, float* __restrict__ z) {
  __shared__ float sm[8];
  const int tok = blockIdx.x, t = threadIdx.x;
  const float4 h = *(const float4*)(hs + tok * 4);
  float xv[4]; float ss = 0.f;
  #pragma unroll
  for (int j = 0; j < 4; ++j) {
    int d = t + j * 256;
    float v = x[tok * DM + d]
            + h.x * Cm[0 * DM + d] + h.y * Cm[1 * DM + d]
            + h.z * Cm[2 * DM + d] + h.w * Cm[3 * DM + d];
    x[tok * DM + d] = v;
    xv[j] = v; ss += v * v;
  }
  float rstd = rsqrtf(block_reduce_256(ss, sm) * (1.0f / DM) + EPSN);
  float acc[8] = {0.f, 0.f, 0.f, 0.f, 0.f, 0.f, 0.f, 0.f};
  #pragma unroll
  for (int j = 0; j < 4; ++j) {
    int d = t + j * 256;
    float xn = xv[j] * nw[d] * rstd;
    #pragma unroll
    for (int r = 0; r < 8; ++r) acc[r] += xn * W1[d * 8 + r];
  }
  #pragma unroll
  for (int r = 0; r < 8; ++r) {
    float tot = block_reduce_256(acc[r], sm);
    if (t == 0) z[tok * 8 + r] = gelu_tanh(tot);
  }
}

// ---- 5) x += z@W2 ; then either next-layer (rmsnorm@Bm -> u) or final norm
//         to split-f16 (hi/lo) activations for the WMMA GEMM. ----------------

__global__ void k_ffn2_norm(float* __restrict__ x, const float* __restrict__ z,
                            const float* __restrict__ W2, const float* __restrict__ nw,
                            const float* __restrict__ Bm, float* __restrict__ u,
                            _Float16* __restrict__ xhh, _Float16* __restrict__ xhl,
                            int mode) {
  __shared__ float sm[8];
  const int tok = blockIdx.x, t = threadIdx.x;
  float zr[8];
  #pragma unroll
  for (int r = 0; r < 8; ++r) zr[r] = z[tok * 8 + r];
  float xv[4]; float ss = 0.f;
  #pragma unroll
  for (int j = 0; j < 4; ++j) {
    int d = t + j * 256;
    float v = x[tok * DM + d];
    #pragma unroll
    for (int r = 0; r < 8; ++r) v += zr[r] * W2[r * DM + d];
    x[tok * DM + d] = v;
    xv[j] = v; ss += v * v;
  }
  float rstd = rsqrtf(block_reduce_256(ss, sm) * (1.0f / DM) + EPSN);
  if (mode == 0) {
    float acc[4] = {0.f, 0.f, 0.f, 0.f};
    #pragma unroll
    for (int j = 0; j < 4; ++j) {
      int d = t + j * 256;
      float xn = xv[j] * nw[d] * rstd;
      #pragma unroll
      for (int r = 0; r < 4; ++r) acc[r] += xn * Bm[d * 4 + r];
    }
    #pragma unroll
    for (int r = 0; r < 4; ++r) {
      float tot = block_reduce_256(acc[r], sm);
      if (t == 0) u[tok * 4 + r] = tot;
    }
  } else {
    #pragma unroll
    for (int j = 0; j < 4; ++j) {
      int d = t + j * 256;
      float xn = xv[j] * nw[d] * rstd;
      _Float16 hi = (_Float16)xn;
      xhh[tok * DM + d] = hi;
      xhl[tok * DM + d] = (_Float16)(xn - (float)hi);
    }
  }
}

// ---------------- 6) logits GEMM: out[2048,50000] = Xn @ Table^T -------------
// Split-f16 3-pass WMMA (AhBh + AhBl + AlBh). Block tile 64Mx128N, 8 waves of
// 32x32, K-step 32. Double-buffered LDS: while tile k feeds the WMMAs, tile
// k+1's A panel arrives via async global->LDS and its B panel is generated
// from the TT cores. One barrier per K-step.

__global__ __launch_bounds__(256) void k_gemm(
    const _Float16* __restrict__ ah, const _Float16* __restrict__ al,
    const float* __restrict__ c1, const float* __restrict__ c2,
    float* __restrict__ out) {
  __shared__ _Float16 AsH[2][BMT * LDA];   // 10 KB
  __shared__ _Float16 AsL[2][BMT * LDA];   // 10 KB
  __shared__ _Float16 BsH[2][BNT * LDA];   // 20 KB
  __shared__ _Float16 BsL[2][BNT * LDA];   // 20 KB

  const int t     = threadIdx.x;
  const int nBase = blockIdx.x * BNT;
  const int mBase = blockIdx.y * BMT;

  // ---- per-thread table-column setup: one vocab column, 16 of the 32 d2 ----
  const int bn  = t & 127;
  const int d2b = (t >> 7) * 16;
  int v = nBase + bn; if (v >= VOC) v = VOC - 1;     // clamp; masked at store
  const int i1 = v / VF2;
  const int i2 = v - i1 * VF2;
  float c2r0[16], c2r1[16];
  #pragma unroll
  for (int j = 0; j < 16; ++j) {
    c2r0[j] = c2[(i2 * 2 + 0) * 32 + d2b + j];
    c2r1[j] = c2[(i2 * 2 + 1) * 32 + d2b + j];
  }
  const float* c1p = c1 + i1 * 64;                   // [d1=32][r=2]

  // ---- A-copy setup: 256 chunks of 16B (hi:128, lo:128), 1 per thread ----
  const int aArr = t >> 7;                // 0 = hi, 1 = lo
  const int aIdx = t & 127;
  const int aRow = aIdx >> 2;
  const int aCol = (aIdx & 3) * 8;

  const int lane = t & 31;
  const int w    = t >> 5;
  const int wm   = w >> 2;          // 0..1 -> M offset wm*32
  const int wn   = w & 3;           // 0..3 -> N offset wn*32
  const int l16  = lane & 15;
  const int lhi  = lane >> 4;       // 0|1

  // stage tile kk into buffer buf (A panel copy + B panel generation)
  auto stage = [&](int kk, int buf) {
    size_t g = (size_t)(mBase + aRow) * DM + kk * 32 + aCol;
    if (aArr == 0) g2l_b128(&ah[g], &AsH[buf][aRow * LDA + aCol]);
    else           g2l_b128(&al[g], &AsL[buf][aRow * LDA + aCol]);
    float a0 = c1p[kk * 2 + 0];
    float a1 = c1p[kk * 2 + 1];
    #pragma unroll
    for (int hseg = 0; hseg < 2; ++hseg) {
      v8h bh, bl;
      #pragma unroll
      for (int j = 0; j < 8; ++j) {
        float val = a0 * c2r0[hseg * 8 + j] + a1 * c2r1[hseg * 8 + j];
        _Float16 hi = (_Float16)val;
        bh[j] = hi;
        bl[j] = (_Float16)(val - (float)hi);
      }
      *(v8h*)&BsH[buf][bn * LDA + d2b + hseg * 8] = bh;
      *(v8h*)&BsL[buf][bn * LDA + d2b + hseg * 8] = bl;
    }
  };

  v8f acc[2][2];
  #pragma unroll
  for (int mf = 0; mf < 2; ++mf)
    #pragma unroll
    for (int nf = 0; nf < 2; ++nf)
      #pragma unroll
      for (int r = 0; r < 8; ++r) acc[mf][nf][r] = 0.f;

  // prologue: tile 0 into buffer 0
  stage(0, 0);
  wait_async0();
  __syncthreads();

  int p = 0;
  for (int kk = 0; kk < 32; ++kk) {                  // K = 1024, step 32 (=d1)
    if (kk < 31) stage(kk + 1, p ^ 1);               // overlaps with compute

    // ---- A fragments (ISA 16-bit A 16x32 layout) ----
    v16h afh[2], afl[2];
    #pragma unroll
    for (int mf = 0; mf < 2; ++mf) {
      int row = (wm * 32 + mf * 16 + l16) * LDA;
      int off = lhi * 8;                              // lanes16-31: K=8..,24..
      v8h x0 = *(const v8h*)&AsH[p][row + off];
      v8h x1 = *(const v8h*)&AsH[p][row + off + 16];
      afh[mf] = __builtin_shufflevector(x0, x1, 0,1,2,3,4,5,6,7,8,9,10,11,12,13,14,15);
      v8h y0 = *(const v8h*)&AsL[p][row + off];
      v8h y1 = *(const v8h*)&AsL[p][row + off + 16];
      afl[mf] = __builtin_shufflevector(y0, y1, 0,1,2,3,4,5,6,7,8,9,10,11,12,13,14,15);
    }
    // ---- B fragments + 3-pass WMMA ----
    #pragma unroll
    for (int nf = 0; nf < 2; ++nf) {
      int col = (wn * 32 + nf * 16 + l16) * LDA;
      int off = lhi * 8;                              // lanes16-31: K=8..,24..
      v8h b0 = *(const v8h*)&BsH[p][col + off];
      v8h b1 = *(const v8h*)&BsH[p][col + off + 16];
      v16h bfh = __builtin_shufflevector(b0, b1, 0,1,2,3,4,5,6,7,8,9,10,11,12,13,14,15);
      v8h e0 = *(const v8h*)&BsL[p][col + off];
      v8h e1 = *(const v8h*)&BsL[p][col + off + 16];
      v16h bfl = __builtin_shufflevector(e0, e1, 0,1,2,3,4,5,6,7,8,9,10,11,12,13,14,15);
      #pragma unroll
      for (int mf = 0; mf < 2; ++mf) {
        acc[mf][nf] = __builtin_amdgcn_wmma_f32_16x16x32_f16(
            false, afh[mf], false, bfh, (short)0, acc[mf][nf], false, false);
        acc[mf][nf] = __builtin_amdgcn_wmma_f32_16x16x32_f16(
            false, afh[mf], false, bfl, (short)0, acc[mf][nf], false, false);
        acc[mf][nf] = __builtin_amdgcn_wmma_f32_16x16x32_f16(
            false, afl[mf], false, bfh, (short)0, acc[mf][nf], false, false);
      }
    }

    wait_async0();          // next tile's A landed in LDS
    __syncthreads();        // + everyone done reading buffer p
    p ^= 1;
  }

  // ---- store C (ISA 32-bit C 16x16 layout: VGPR r -> M=r / M=8+r) ----
  #pragma unroll
  for (int nf = 0; nf < 2; ++nf) {
    int col = nBase + wn * 32 + nf * 16 + l16;
    if (col < VOC) {
      #pragma unroll
      for (int mf = 0; mf < 2; ++mf) {
        int row0 = mBase + wm * 32 + mf * 16 + lhi * 8;
        #pragma unroll
        for (int r = 0; r < 8; ++r)
          __builtin_nontemporal_store(acc[mf][nf][r],
                                      &out[(size_t)(row0 + r) * VOC + col]);
      }
    }
  }
}

// ---------------------------------------------------------------------------

extern "C" void kernel_launch(void* const* d_in, const int* in_sizes, int n_in,
                              void* d_out, int out_size, void* d_ws, size_t ws_size,
                              hipStream_t stream) {
  (void)in_sizes; (void)n_in; (void)out_size; (void)ws_size;
  const int*   ids     = (const int*)d_in[0];
  const float* core1   = (const float*)d_in[1];   // [250,32,2]
  const float* core2   = (const float*)d_in[2];   // [200,2,32]
  const float* norm_w  = (const float*)d_in[3];   // [4,1024]
  const float* a_decay = (const float*)d_in[4];   // [4,4]
  const float* Bm      = (const float*)d_in[5];   // [4,1024,4]
  const float* Cm      = (const float*)d_in[6];   // [4,4,1024]
  const float* W1      = (const float*)d_in[7];   // [4,1024,8]
  const float* W2      = (const float*)d_in[8];   // [4,8,1024]
  const float* fw      = (const float*)d_in[9];   // [1024]
  float* out = (float*)d_out;

  // workspace carve (~16.8 MB)
  float*    x   = (float*)d_ws;                       // 2048*1024 f32
  float*    u   = x + (size_t)NTOK * DM;              // 2048*4    f32
  float*    z   = u + (size_t)NTOK * 4;               // 2048*8    f32
  _Float16* xhh = (_Float16*)(z + (size_t)NTOK * 8);  // 2048*1024 f16 (hi)
  _Float16* xhl = xhh + (size_t)NTOK * DM;            // 2048*1024 f16 (lo)

  k_embed<<<NTOK, 256, 0, stream>>>(ids, core1, core2, x);
  k_norm_B<<<NTOK, 256, 0, stream>>>(x, norm_w, Bm, u);
  for (int l = 0; l < NLAY; ++l) {
    k_scan<<<1, 32, 0, stream>>>(u, a_decay + l * 4);
    k_cy_ffn1<<<NTOK, 256, 0, stream>>>(x, u, Cm + (size_t)l * 4 * DM,
                                        norm_w + (size_t)l * DM,
                                        W1 + (size_t)l * DM * 8, z);
    if (l < NLAY - 1) {
      k_ffn2_norm<<<NTOK, 256, 0, stream>>>(x, z, W2 + (size_t)l * 8 * DM,
                                            norm_w + (size_t)(l + 1) * DM,
                                            Bm + (size_t)(l + 1) * DM * 4, u,
                                            nullptr, nullptr, 0);
    } else {
      k_ffn2_norm<<<NTOK, 256, 0, stream>>>(x, z, W2 + (size_t)l * 8 * DM,
                                            fw, nullptr, nullptr,
                                            xhh, xhl, 1);
    }
  }
  dim3 grid((VOC + BNT - 1) / BNT, NTOK / BMT);
  k_gemm<<<grid, 256, 0, stream>>>(xhh, xhl, core1, core2, out);
}
